// BruteForceRetriever_62818191671480
// MI455X (gfx1250) — matrix operand order repaired and checked by reading, not compiled, not run on hardware
//
#include <hip/hip_runtime.h>
#include <hip/hip_bf16.h>
#include <math.h>

// ---------------------------------------------------------------------------
// BruteForceRetriever for MI455X (gfx1250, wave32, WMMA)
//   sim = normalize(Q) @ bankT   (bank pre-normalized)
//   top-8 per query, gather passage tokens
// ---------------------------------------------------------------------------

typedef __bf16 bf16_t;
typedef __attribute__((ext_vector_type(16))) __bf16 v16bf;
typedef __attribute__((ext_vector_type(8)))  __bf16 v8bf;
typedef __attribute__((ext_vector_type(8)))  float  v8f;

#define EMBED   512
#define NQ      256
#define TOPK    8
#define PLEN    16
#define NB      128      // passages per LDS tile
#define NBP     132      // padded score-row stride (bank-conflict avoidance)
#define NBLOCKS 512      // passage-chunk blocks

// ---------------------------------------------------------------------------
// top-8 insertion: array kept ascending by (score, -index) lexicographic key.
// All indices compile-time constant -> stays in VGPRs.
// ---------------------------------------------------------------------------
__device__ __forceinline__ void topk_insert(float (&bs)[TOPK], unsigned (&bi)[TOPK],
                                            float s, unsigned idx) {
    bool better = (s > bs[0]) || (s == bs[0] && idx < bi[0]);
    if (!better) return;
    bs[0] = s; bi[0] = idx;
#pragma unroll
    for (int k = 0; k < TOPK - 1; ++k) {
        bool sw = (bs[k] > bs[k + 1]) || (bs[k] == bs[k + 1] && bi[k] < bi[k + 1]);
        float  ts = sw ? bs[k + 1] : bs[k];
        float  tS = sw ? bs[k]     : bs[k + 1];
        unsigned ti = sw ? bi[k + 1] : bi[k];
        unsigned tI = sw ? bi[k]     : bi[k + 1];
        bs[k] = ts; bs[k + 1] = tS;
        bi[k] = ti; bi[k + 1] = tI;
    }
}

// ---------------------------------------------------------------------------
// Kernel 0: L2-normalize queries (matches F.normalize p=2, eps=1e-12),
//           emit bf16 copy for the WMMA A-operand. One block per query.
// ---------------------------------------------------------------------------
__global__ void __launch_bounds__(128)
normalize_queries_kernel(const float* __restrict__ q, bf16_t* __restrict__ qb) {
    const int query = blockIdx.x;
    const int t = threadIdx.x;
    const float* row = q + (size_t)query * EMBED;

    float vals[4];
    float ss = 0.f;
#pragma unroll
    for (int i = 0; i < 4; ++i) {
        vals[i] = row[t + 128 * i];
        ss += vals[i] * vals[i];
    }
    // wave32 reduction then cross-wave via LDS
#pragma unroll
    for (int off = 16; off > 0; off >>= 1) ss += __shfl_xor(ss, off, 32);
    __shared__ float red[4];
    if ((t & 31) == 0) red[t >> 5] = ss;
    __syncthreads();
    float total = red[0] + red[1] + red[2] + red[3];
    float nrm = sqrtf(total);
    float denom = fmaxf(nrm, 1e-12f);
    float scale = 1.0f / denom;
#pragma unroll
    for (int i = 0; i < 4; ++i)
        qb[(size_t)query * EMBED + t + 128 * i] = (bf16_t)(vals[i] * scale);
}

// ---------------------------------------------------------------------------
// Kernel 1: blocked MIPS + per-block top-8.
//  - block owns a contiguous passage chunk, iterates NB=128-row tiles
//  - fp32 tile -> bf16 LDS (converted once, reused by all 16 M-tiles)
//  - 8 waves x (2 M-tiles x 8 N-subtiles) of v_wmma_f32_16x16x32_bf16
//  - scores to LDS, thread q keeps running top-8 of its query in registers
// ---------------------------------------------------------------------------
__global__ void __launch_bounds__(256)
retrieve_score_kernel(const float* __restrict__ bank,
                      const bf16_t* __restrict__ qb,
                      float* __restrict__ cand,
                      int npass, int chunk) {
    extern __shared__ __align__(32) char smem[];
    bf16_t* sB = (bf16_t*)smem;                          // [NB][EMBED] bf16
    float*  sS = (float*)(smem + (size_t)NB * EMBED * 2);// [NQ][NBP]  f32

    const int tid  = threadIdx.x;
    const int lane = tid & 31;
    const int wave = tid >> 5;
    const int half = lane >> 4;   // 0 or 1
    const int mrow = lane & 15;

    const int n_begin = blockIdx.x * chunk;
    const int n_end   = min(npass, n_begin + chunk);

    float    best_s[TOPK];
    unsigned best_i[TOPK];
#pragma unroll
    for (int k = 0; k < TOPK; ++k) { best_s[k] = -__builtin_inff(); best_i[k] = 0xFFFFFFFFu; }

    const int m0 = wave * 32;     // queries [m0, m0+32) for this wave
    const v8f vzero = {0.f, 0.f, 0.f, 0.f, 0.f, 0.f, 0.f, 0.f};

    for (int tile_n = n_begin; tile_n < n_end; tile_n += NB) {
        // ---- stage: bank fp32 -> bf16 LDS (float4 per thread-iter) ----
#pragma unroll
        for (int it = 0; it < (NB * EMBED / 4) / 256; ++it) {
            int e4 = tid + 256 * it;
            int e  = e4 * 4;
            int r  = e >> 9;          // row in tile
            int c  = e & (EMBED - 1); // col
            int n  = tile_n + r;
            float4 v = make_float4(0.f, 0.f, 0.f, 0.f);
            if (n < npass) {
                v = *(const float4*)(bank + (size_t)n * EMBED + c);
                if (n + NB < npass)   // warm next tile into cache: global_prefetch_b8
                    __builtin_prefetch(bank + (size_t)(n + NB) * EMBED + c, 0, 1);
            }
            sB[r * EMBED + c + 0] = (bf16_t)v.x;
            sB[r * EMBED + c + 1] = (bf16_t)v.y;
            sB[r * EMBED + c + 2] = (bf16_t)v.z;
            sB[r * EMBED + c + 3] = (bf16_t)v.w;
        }
        __syncthreads();

        // ---- compute: 2 M-tiles x 8 N-subtiles, K-loop over 512 ----
        v8f acc[2][8];
#pragma unroll
        for (int mt = 0; mt < 2; ++mt)
#pragma unroll
            for (int s = 0; s < 8; ++s) acc[mt][s] = vzero;

        for (int ks = 0; ks < EMBED; ks += 32) {
            // A-operand (16x32, contiguous k order shared with B): lane m holds K {16h..16h+15}
            v16bf a[2];
#pragma unroll
            for (int mt = 0; mt < 2; ++mt) {
                const bf16_t* ap = qb + (size_t)(m0 + mt * 16 + mrow) * EMBED + ks + 16 * half;
                union { v16bf v; v8bf h[2]; } u;
                u.h[0] = *(const v8bf*)(ap);
                u.h[1] = *(const v8bf*)(ap + 8);
                a[mt] = u.v;
            }
#pragma unroll
            for (int s = 0; s < 8; ++s) {
                // B-operand (32x16): lanes 0-15 N=lane K=ks..ks+15; lanes 16-31 K=ks+16..+31
                const bf16_t* bp = sB + (size_t)(s * 16 + mrow) * EMBED + ks + 16 * half;
                v16bf b = *(const v16bf*)bp;
                acc[0][s] = __builtin_amdgcn_wmma_f32_16x16x32_bf16(
                    false, a[0], false, b, (short)0, acc[0][s], false, false);
                acc[1][s] = __builtin_amdgcn_wmma_f32_16x16x32_bf16(
                    false, a[1], false, b, (short)0, acc[1][s], false, false);
            }
        }

        // ---- C layout: VGPR r, lanes 0-15 M=r, lanes 16-31 M=r+8; N=lane%16 ----
#pragma unroll
        for (int mt = 0; mt < 2; ++mt)
#pragma unroll
            for (int s = 0; s < 8; ++s)
#pragma unroll
                for (int r = 0; r < 8; ++r) {
                    int query = m0 + mt * 16 + r + 8 * half;
                    int ncol  = s * 16 + mrow;
                    sS[query * NBP + ncol] = acc[mt][s][r];
                }
        __syncthreads();

        // ---- per-query running top-8 (thread tid <-> query tid) ----
        int limit = min(n_end - tile_n, NB);
        for (int j = 0; j < limit; ++j)
            topk_insert(best_s, best_i, sS[tid * NBP + j], (unsigned)(tile_n + j));
        // (no barrier needed before next staging: staging writes sB only,
        //  and next compute is fenced by the post-staging barrier)
    }

    // ---- emit block-local candidates, descending ----
    float* out = cand + ((size_t)tid * NBLOCKS + blockIdx.x) * (2 * TOPK);
#pragma unroll
    for (int k = 0; k < TOPK; ++k) {
        out[2 * k] = best_s[TOPK - 1 - k];
        ((unsigned*)out)[2 * k + 1] = best_i[TOPK - 1 - k];
    }
}

// ---------------------------------------------------------------------------
// Kernel 2: merge NBLOCKS*8 candidates per query -> global top-8,
//           write scores + gathered tokens (as float, per harness d_out).
// ---------------------------------------------------------------------------
__global__ void __launch_bounds__(256)
reduce_topk_kernel(const float* __restrict__ cand,
                   const int* __restrict__ tokens,
                   float* __restrict__ outp) {
    const int q = blockIdx.x;
    const int t = threadIdx.x;
    const float* cq = cand + (size_t)q * NBLOCKS * (2 * TOPK);
    const int total = NBLOCKS * TOPK;

    float    bs[TOPK];
    unsigned bi[TOPK];
#pragma unroll
    for (int k = 0; k < TOPK; ++k) { bs[k] = -__builtin_inff(); bi[k] = 0xFFFFFFFFu; }

    for (int e = t; e < total; e += 256) {
        float    s   = cq[2 * e];
        unsigned idx = ((const unsigned*)cq)[2 * e + 1];
        topk_insert(bs, bi, s, idx);
    }

    __shared__ float    ls[256 * TOPK];
    __shared__ unsigned li[256 * TOPK];
#pragma unroll
    for (int k = 0; k < TOPK; ++k) { ls[t * TOPK + k] = bs[k]; li[t * TOPK + k] = bi[k]; }
    __syncthreads();

    for (int stride = 128; stride >= 1; stride >>= 1) {
        if (t < stride) {
#pragma unroll
            for (int k = 0; k < TOPK; ++k)
                topk_insert(bs, bi, ls[(t + stride) * TOPK + k], li[(t + stride) * TOPK + k]);
#pragma unroll
            for (int k = 0; k < TOPK; ++k) { ls[t * TOPK + k] = bs[k]; li[t * TOPK + k] = bi[k]; }
        }
        __syncthreads();
    }

    __shared__ unsigned topidx[TOPK];
    if (t == 0) {
#pragma unroll
        for (int k = 0; k < TOPK; ++k) {          // descending like jax top_k
            outp[NQ * TOPK * PLEN + q * TOPK + k] = bs[TOPK - 1 - k];
            topidx[k] = bi[TOPK - 1 - k];
        }
    }
    __syncthreads();

    if (t < TOPK * PLEN) {                        // 128 gathered tokens
        unsigned idx = topidx[t >> 4];
        outp[(size_t)q * (TOPK * PLEN) + t] = (float)tokens[(size_t)idx * PLEN + (t & 15)];
    }
}

// ---------------------------------------------------------------------------
extern "C" void kernel_launch(void* const* d_in, const int* in_sizes, int n_in,
                              void* d_out, int out_size, void* d_ws, size_t ws_size,
                              hipStream_t stream) {
    const float* q_embeds = (const float*)d_in[0];
    const float* bank     = (const float*)d_in[1];
    const int*   tokens   = (const int*)d_in[2];
    // d_in[3] = top_k scalar (fixed at 8 per reference)
    float* outp = (float*)d_out;

    const int npass = in_sizes[1] / EMBED;
    const int chunk = (npass + NBLOCKS - 1) / NBLOCKS;

    // workspace: [candidates: NQ*NBLOCKS*8*(score,idx)] [bf16 queries]
    char* ws = (char*)d_ws;
    float*  cand = (float*)ws;
    bf16_t* qb   = (bf16_t*)(ws + (size_t)NQ * NBLOCKS * TOPK * 2 * sizeof(float));

    normalize_queries_kernel<<<NQ, 128, 0, stream>>>(q_embeds, qb);

    size_t smem = (size_t)NB * EMBED * sizeof(bf16_t)   // 128 KB bf16 tile
                + (size_t)NQ * NBP * sizeof(float);     // 132 KB score buffer
    retrieve_score_kernel<<<NBLOCKS, 256, smem, stream>>>(bank, qb, cand, npass, chunk);

    reduce_topk_kernel<<<NQ, 256, 0, stream>>>(cand, tokens, outp);

    (void)n_in; (void)out_size; (void)ws_size;
}
